// Breadth_6408091205708
// MI455X (gfx1250) — matrix operand order repaired and checked by reading, not compiled, not run on hardware
//
#include <hip/hip_runtime.h>
#include <hip/hip_bf16.h>

typedef __attribute__((ext_vector_type(16))) _Float16 v16h;
typedef __attribute__((ext_vector_type(8)))  _Float16 v8h;
typedef __attribute__((ext_vector_type(8)))  float    v8f;

#define NSLOPE 0.2f
#define EPSF   1e-16f

// ---------------- init: zero accumulator (f32x4) & denom, m = -inf ----------
__global__ __launch_bounds__(256) void k_init(float4* acc4, float* denom, float* mbuf, int N) {
    int i = blockIdx.x * blockDim.x + threadIdx.x;
    if (i < N * 16) acc4[i] = make_float4(0.f, 0.f, 0.f, 0.f);
    if (i < N) { denom[i] = 0.0f; mbuf[i] = -__builtin_inff(); }
}

// ---------------- prologue: x (f32) -> xh (f16) -----------------------------
__global__ __launch_bounds__(256) void k_cvt_x(const float* __restrict__ x,
                                               _Float16* __restrict__ xh, int total) {
    int i = blockIdx.x * blockDim.x + threadIdx.x;
    if (i < total) xh[i] = (_Float16)x[i];
}

// ---------------- prologue: W (f32, [k][n]) -> wt (f16, [n][k]) -------------
__global__ __launch_bounds__(256) void k_cvt_w(const float* __restrict__ W,
                                               _Float16* __restrict__ wt) {
    int i = blockIdx.x * blockDim.x + threadIdx.x;   // i = n*64 + k
    if (i < 64 * 64) {
        int n = i >> 6, k = i & 63;
        wt[i] = (_Float16)W[k * 64 + n];
    }
}

// ---------------- h = x @ W via v_wmma_f32_16x16x32_f16 ---------------------
// One wave32 per 16-row tile; fragments loaded as b128 f16 vectors; D tile
// transposed through wave-private LDS for coalesced b128 stores.
__global__ __launch_bounds__(256) void k_gemm_wmma(const _Float16* __restrict__ xh,
                                                   const _Float16* __restrict__ wt,
                                                   float* __restrict__ h, int N) {
    __shared__ float lds[8][16 * 68];                // 4.25KB per wave, pitch 68 deskews banks
    int gid  = blockIdx.x * blockDim.x + threadIdx.x;
    int wave = gid >> 5;
    int lane = threadIdx.x & 31;
    int wib  = (threadIdx.x >> 5);
    int tiles = (N + 15) >> 4;
    if (wave >= tiles) return;

    int row_base = wave << 4;
    int m    = lane & 15;                            // M (A) / N (B,C,D)
    int half = lane >> 4;                            // K sub-range selector
    int row  = row_base + m;
    int rowc = (row < N) ? row : (N - 1);            // branchless clamp for loads

    v8f acc[4] = {};

    #pragma unroll
    for (int kb = 0; kb < 64; kb += 32) {
        // A 16x32 f16 frag: elements 0..7 => K = kb+half*8+e ; 8..15 => +16
        const _Float16* ar = xh + (size_t)rowc * 64 + kb + half * 8;
        v8h alo = *(const v8h*)(ar);
        v8h ahi = *(const v8h*)(ar + 16);
        v16h a = __builtin_shufflevector(alo, ahi,
                 0,1,2,3,4,5,6,7,8,9,10,11,12,13,14,15);
        #pragma unroll
        for (int t = 0; t < 4; ++t) {
            // B 32x16 f16 frag from transposed W: elements 0..7 => K = kb+half*8+e ; 8..15 => +16
            int n = t * 16 + m;
            const _Float16* br = wt + (size_t)n * 64 + kb + half * 8;
            v8h blo = *(const v8h*)(br);
            v8h bhi = *(const v8h*)(br + 16);
            v16h b = __builtin_shufflevector(blo, bhi,
                     0,1,2,3,4,5,6,7,8,9,10,11,12,13,14,15);
            acc[t] = __builtin_amdgcn_wmma_f32_16x16x32_f16(
                false, a, false, b, (short)0, acc[t], false, false);
        }
    }

    // D: VGPR i holds (M = i + half*8, N = t*16 + m)  ->  LDS transpose
    #pragma unroll
    for (int t = 0; t < 4; ++t) {
        #pragma unroll
        for (int i = 0; i < 8; ++i)
            lds[wib][(i + half * 8) * 68 + t * 16 + m] = acc[t][i];
    }
    // coalesced write-back: 256 float4 quads / 32 lanes = 8 per lane
    #pragma unroll
    for (int it = 0; it < 8; ++it) {
        int q  = it * 32 + lane;
        int r  = q >> 4;                              // row within tile
        int qc = q & 15;                              // quad within row
        int gr = row_base + r;
        if (gr < N) {
            float4 v = *(float4*)&lds[wib][r * 68 + qc * 4];
            *(float4*)&h[(size_t)gr * 64 + qc * 4] = v;
        }
    }
}

// ---------------- per-node attention logits: wave per node, shfl reduce -----
__global__ __launch_bounds__(256) void k_att(const float* __restrict__ h,
                                             const float* __restrict__ att_src,
                                             const float* __restrict__ att_dst,
                                             float* a_src, float* a_dst, int N) {
    int gid  = blockIdx.x * blockDim.x + threadIdx.x;
    int node = gid >> 5;
    int lane = threadIdx.x & 31;
    if (node >= N) return;
    const float* hr = h + (size_t)node * 64;
    float2 hv = *(const float2*)(hr + 2 * lane);
    float s = hv.x * att_src[2 * lane] + hv.y * att_src[2 * lane + 1];
    float d = hv.x * att_dst[2 * lane] + hv.y * att_dst[2 * lane + 1];
    #pragma unroll
    for (int off = 16; off > 0; off >>= 1) {
        s += __shfl_xor(s, off, 32);
        d += __shfl_xor(d, off, 32);
    }
    if (lane == 0) { a_src[node] = s; a_dst[node] = d; }
}

// float atomic max via monotone integer atomics (works with -inf init)
__device__ inline void atomicMaxF(float* addr, float val) {
    if (val >= 0.0f) atomicMax((int*)addr, __float_as_int(val));
    else             atomicMin((unsigned int*)addr, __float_as_uint(val));
}

// ---------------- edge pass 1: leaky_relu logit + segment max over dst ------
__global__ __launch_bounds__(256) void k_edge_max(const long long* __restrict__ ei,
                                                  const float* __restrict__ a_src,
                                                  const float* __restrict__ a_dst,
                                                  float* __restrict__ ebuf,
                                                  float* __restrict__ mbuf, int E) {
    int e = blockIdx.x * blockDim.x + threadIdx.x;
    if (e >= E) return;
    int s = (int)ei[e];
    int d = (int)ei[(size_t)E + e];
    float v = a_src[s] + a_dst[d];
    v = (v > 0.0f) ? v : NSLOPE * v;
    ebuf[e] = v;
    atomicMaxF(&mbuf[d], v);
}

// ---------------- edge pass 2: exp(e - m[dst]) + segment sum ----------------
__global__ __launch_bounds__(256) void k_edge_sum(const long long* __restrict__ ei,
                                                  float* __restrict__ ebuf,
                                                  const float* __restrict__ mbuf,
                                                  float* __restrict__ denom, int E) {
    int e = blockIdx.x * blockDim.x + threadIdx.x;
    if (e >= E) return;
    int d = (int)ei[(size_t)E + e];
    float v = __expf(ebuf[e] - mbuf[d]);
    ebuf[e] = v;
    atomicAdd(&denom[d], v);
}

// ---------------- edge pass 3: scatter alpha * h[src] into acc[dst] ---------
// One wave per edge; each lane handles channels {2*lane, 2*lane+1} (b64 read).
__global__ __launch_bounds__(256) void k_scatter(const long long* __restrict__ ei,
                                                 const float* __restrict__ h,
                                                 const float* __restrict__ ebuf,
                                                 const float* __restrict__ denom,
                                                 float* __restrict__ acc, int E) {
    int gid  = blockIdx.x * blockDim.x + threadIdx.x;
    int e    = gid >> 5;
    int lane = threadIdx.x & 31;
    if (e >= E) return;
    int s = (int)ei[e];
    int d = (int)ei[(size_t)E + e];
    float alpha = ebuf[e] / (denom[d] + EPSF);
    float2 hv = *(const float2*)(h + (size_t)s * 64 + 2 * lane);
    float* ad = acc + (size_t)d * 64 + 2 * lane;
    atomicAdd(ad,     alpha * hv.x);
    atomicAdd(ad + 1, alpha * hv.y);
}

// ---------------- finalize: tanh(acc + bias) -> out (float4) ----------------
__global__ __launch_bounds__(256) void k_final(const float4* __restrict__ acc4,
                                               const float4* __restrict__ bias4,
                                               float4* __restrict__ out4, int N) {
    int i = blockIdx.x * blockDim.x + threadIdx.x;
    if (i >= N * 16) return;
    float4 a = acc4[i];
    float4 b = bias4[i & 15];
    out4[i] = make_float4(tanhf(a.x + b.x), tanhf(a.y + b.y),
                          tanhf(a.z + b.z), tanhf(a.w + b.w));
}

extern "C" void kernel_launch(void* const* d_in, const int* in_sizes, int n_in,
                              void* d_out, int out_size, void* d_ws, size_t ws_size,
                              hipStream_t stream) {
    const float*     x       = (const float*)d_in[0];
    const long long* ei      = (const long long*)d_in[1];   // int64 [2, E]
    const float*     W       = (const float*)d_in[2];
    const float*     att_src = (const float*)d_in[3];
    const float*     att_dst = (const float*)d_in[4];
    const float*     bias    = (const float*)d_in[5];
    float*           out     = (float*)d_out;

    const int N = in_sizes[0] / 64;
    const int E = in_sizes[1] / 2;

    // workspace layout
    float* ws     = (float*)d_ws;
    float* h      = ws;                         // N*64 f32
    float* acc    = h      + (size_t)N * 64;    // N*64 f32
    float* a_src  = acc    + (size_t)N * 64;    // N
    float* a_dst  = a_src  + N;                 // N
    float* mbuf   = a_dst  + N;                 // N
    float* denom  = mbuf   + N;                 // N
    float* ebuf   = denom  + N;                 // E
    _Float16* xh  = (_Float16*)(ebuf + E);      // N*64 f16
    _Float16* wt  = xh + (size_t)N * 64;        // 64*64 f16 (W transposed)

    const int B = 256;
    int g_init   = (N * 16 + B - 1) / B;
    int g_cvtx   = (N * 64 + B - 1) / B;
    int g_cvtw   = (64 * 64 + B - 1) / B;
    int g_gemm   = (((N + 15) / 16) * 32 + B - 1) / B;
    int g_att    = (int)(((size_t)N * 32 + B - 1) / B);
    int g_edge   = (E + B - 1) / B;
    int g_scat   = (int)(((size_t)E * 32 + B - 1) / B);
    int g_final  = (N * 16 + B - 1) / B;

    k_init     <<<g_init,  B, 0, stream>>>((float4*)acc, denom, mbuf, N);
    k_cvt_x    <<<g_cvtx,  B, 0, stream>>>(x, xh, N * 64);
    k_cvt_w    <<<g_cvtw,  B, 0, stream>>>(W, wt);
    k_gemm_wmma<<<g_gemm,  B, 0, stream>>>(xh, wt, h, N);
    k_att      <<<g_att,   B, 0, stream>>>(h, att_src, att_dst, a_src, a_dst, N);
    k_edge_max <<<g_edge,  B, 0, stream>>>(ei, a_src, a_dst, ebuf, mbuf, E);
    k_edge_sum <<<g_edge,  B, 0, stream>>>(ei, ebuf, mbuf, denom, E);
    k_scatter  <<<g_scat,  B, 0, stream>>>(ei, h, ebuf, denom, acc, E);
    k_final    <<<g_final, B, 0, stream>>>((const float4*)acc, (const float4*)bias, (float4*)out, N);
}
